// HIPPI_41669772706343
// MI455X (gfx1250) — matrix operand order repaired and checked by reading, not compiled, not run on hardware
//
#include <hip/hip_runtime.h>

// ---------------------------------------------------------------------------
// Types / helpers
// ---------------------------------------------------------------------------
typedef __bf16 v16bf __attribute__((ext_vector_type(16)));
typedef float  v8f   __attribute__((ext_vector_type(8)));
typedef int    v4i   __attribute__((__vector_size__(16)));   // b128 payload type

union Frag16 {
  uint4 u[2];
  v16bf v;
};

#define MINF 1e-15f
#define PROJ_MAX 0.996f

#define AS1 __attribute__((address_space(1)))
#define AS3 __attribute__((address_space(3)))

#if defined(__AMDGCN__) && __has_builtin(__builtin_amdgcn_global_load_async_to_lds_b128) && \
    __has_builtin(__builtin_amdgcn_s_wait_asynccnt)
#define HGNN_ASYNC 1
#else
#define HGNN_ASYNC 0
#endif

__device__ __forceinline__ unsigned short f2bf(float f) {
  unsigned int x = __float_as_uint(f);
  unsigned int r = x + 0x7FFFu + ((x >> 16) & 1u);
  return (unsigned short)(r >> 16);
}
__device__ __forceinline__ float bf2f(unsigned short u) {
  return __uint_as_float(((unsigned int)u) << 16);
}
__device__ __forceinline__ float waveSum(float v) {
#pragma unroll
  for (int off = 16; off > 0; off >>= 1) v += __shfl_xor(v, off, 32);
  return v;
}
__device__ __forceinline__ float artanh_c(float x) {
  x = fminf(fmaxf(x, -1.0f + 1e-7f), 1.0f - 1e-7f);
  return 0.5f * (log1pf(x) - log1pf(-x));
}

// ---------------------------------------------------------------------------
// WMMA bf16 GEMM:  C[M,Nout] = A[M,K] * W[Nout,K]^T  (+bias, fused epilogue)
// Block tile 128x128, K-step 32, 256 threads = 8 waves (2x4), each wave 4x2
// 16x16 WMMA accumulators. Double-buffered LDS fed by async global->LDS
// loads (ASYNCcnt) when available; synchronous staging otherwise.
// ---------------------------------------------------------------------------
#define TILE_M 128
#define TILE_N 128
#define TILE_K 32
#define LDSS   40   // padded row stride in bf16 elements (80 bytes, 16B aligned)

enum { EPI_F32 = 0, EPI_F32_SIG = 1, EPI_BF16 = 2, EPI_BF16_RELU = 3, EPI_BF16_GATE_RELU = 4 };

__global__ __launch_bounds__(256)
void hgnn_gemm_bf16(const unsigned short* __restrict__ A,
                    const unsigned short* __restrict__ W,
                    const float* __restrict__ bias,
                    int M, int K, int Nout, int epi,
                    float* __restrict__ Cf, unsigned short* __restrict__ Cb,
                    const float* __restrict__ gate, int ldc, int colOff) {
  __shared__ __align__(16) unsigned short shA[2][TILE_M * LDSS];
  __shared__ __align__(16) unsigned short shB[2][TILE_N * LDSS];

  const int tid  = threadIdx.x;
  const int lane = tid & 31;
  const int wid  = tid >> 5;
  const int wm   = (wid >> 2) * 64;  // wave row offset in tile
  const int wn   = (wid & 3) * 32;   // wave col offset in tile
  const int bm   = blockIdx.y * TILE_M;
  const int bn   = blockIdx.x * TILE_N;
  const int ml   = lane & 15;
  const int hi   = lane >> 4;

  // ---- staging coordinates: each thread owns 4x 16B chunks per stage ----
  // OOB rows are clamped (not skipped) so every wave issues exactly 4 async
  // loads per stage -> ASYNCcnt bookkeeping is exact. Clamped rows only feed
  // C rows/cols that the epilogue masks out.
  const int lrow = tid >> 2;         // 0..63
  const int seg  = (tid & 3) * 8;    // 0,8,16,24 (bf16 elems)
  const int grA0 = min(bm + lrow, M - 1);
  const int grA1 = min(bm + lrow + 64, M - 1);
  const int gcB0 = min(bn + lrow, Nout - 1);
  const int gcB1 = min(bn + lrow + 64, Nout - 1);
  const unsigned short* baseA0 = A + (size_t)grA0 * K + seg;
  const unsigned short* baseA1 = A + (size_t)grA1 * K + seg;
  const unsigned short* baseB0 = W + (size_t)gcB0 * K + seg;
  const unsigned short* baseB1 = W + (size_t)gcB1 * K + seg;

  auto issue = [&](int s, int k0) {
#if HGNN_ASYNC
    __builtin_amdgcn_global_load_async_to_lds_b128(
        (AS1 v4i*)(baseA0 + k0), (AS3 v4i*)&shA[s][lrow * LDSS + seg], 0, 0);
    __builtin_amdgcn_global_load_async_to_lds_b128(
        (AS1 v4i*)(baseA1 + k0), (AS3 v4i*)&shA[s][(lrow + 64) * LDSS + seg], 0, 0);
    __builtin_amdgcn_global_load_async_to_lds_b128(
        (AS1 v4i*)(baseB0 + k0), (AS3 v4i*)&shB[s][lrow * LDSS + seg], 0, 0);
    __builtin_amdgcn_global_load_async_to_lds_b128(
        (AS1 v4i*)(baseB1 + k0), (AS3 v4i*)&shB[s][(lrow + 64) * LDSS + seg], 0, 0);
    __builtin_prefetch(baseA0 + k0 + 2 * TILE_K, 0, 1);  // global_prefetch_b8
    __builtin_prefetch(baseB0 + k0 + 2 * TILE_K, 0, 1);
#else
    *(uint4*)&shA[s][lrow * LDSS + seg]        = *(const uint4*)(baseA0 + k0);
    *(uint4*)&shA[s][(lrow + 64) * LDSS + seg] = *(const uint4*)(baseA1 + k0);
    *(uint4*)&shB[s][lrow * LDSS + seg]        = *(const uint4*)(baseB0 + k0);
    *(uint4*)&shB[s][(lrow + 64) * LDSS + seg] = *(const uint4*)(baseB1 + k0);
#endif
  };

  const v8f vzero = {0.f, 0.f, 0.f, 0.f, 0.f, 0.f, 0.f, 0.f};
  v8f acc[4][2];
#pragma unroll
  for (int mi = 0; mi < 4; ++mi)
#pragma unroll
    for (int ni = 0; ni < 2; ++ni) acc[mi][ni] = vzero;

  const int KT = K / TILE_K;
  issue(0, 0);
  for (int kt = 0; kt < KT; ++kt) {
    const int cur = kt & 1;
    if (kt + 1 < KT) {
      issue(cur ^ 1, (kt + 1) * TILE_K);       // overlap next stage with compute
#if HGNN_ASYNC
      __builtin_amdgcn_s_wait_asynccnt(4);     // current stage landed (in-order)
#endif
    } else {
#if HGNN_ASYNC
      __builtin_amdgcn_s_wait_asynccnt(0);
#endif
    }
    __syncthreads();

    // ---- fragments per documented wave32 WMMA VGPR layouts
    Frag16 fa[4], fb[2];
    const int kA = hi * 8;   // lanes 0-15: K 0..7 / 16..23 ; lanes 16-31: K 8..15 / 24..31
    const int kB = hi * 8;   // lanes 0-15: K 0..7 / 16..23 ; lanes 16-31: K 8..15 / 24..31
#pragma unroll
    for (int mi = 0; mi < 4; ++mi) {
      const unsigned short* p = &shA[cur][(wm + mi * 16 + ml) * LDSS];
      fa[mi].u[0] = *(const uint4*)(p + kA);
      fa[mi].u[1] = *(const uint4*)(p + 16 + kA);
    }
#pragma unroll
    for (int ni = 0; ni < 2; ++ni) {
      const unsigned short* p = &shB[cur][(wn + ni * 16 + ml) * LDSS];
      fb[ni].u[0] = *(const uint4*)(p + kB);
      fb[ni].u[1] = *(const uint4*)(p + 16 + kB);
    }
#pragma unroll
    for (int mi = 0; mi < 4; ++mi)
#pragma unroll
      for (int ni = 0; ni < 2; ++ni)
        acc[mi][ni] = __builtin_amdgcn_wmma_f32_16x16x32_bf16(
            false, fa[mi].v, false, fb[ni].v, (short)0, acc[mi][ni], false, false);
    __syncthreads();  // all waves done reading stage before it is overwritten
  }

  // ---- epilogue: C/D layout: VGPR j -> M = j (lanes 0-15) / j+8 (lanes 16-31), N = lane%16
#pragma unroll
  for (int mi = 0; mi < 4; ++mi) {
#pragma unroll
    for (int ni = 0; ni < 2; ++ni) {
#pragma unroll
      for (int j = 0; j < 8; ++j) {
        int rl = wm + mi * 16 + ((lane < 16) ? j : j + 8);
        int cl = wn + ni * 16 + ml;
        int gr = bm + rl, gc = bn + cl;
        if (gr < M && gc < Nout) {
          float v = acc[mi][ni][j];
          if (bias) v += bias[gc];
          size_t di = (size_t)gr * ldc + colOff + gc;
          switch (epi) {
            case EPI_F32:      Cf[di] = v; break;
            case EPI_F32_SIG:  Cf[di] = 1.f / (1.f + expf(-v)); break;
            case EPI_BF16:     Cb[di] = f2bf(v); break;
            case EPI_BF16_RELU:Cb[di] = f2bf(fmaxf(v, 0.f)); break;
            case EPI_BF16_GATE_RELU:
              Cb[di] = f2bf(gate[(size_t)gr * Nout + gc] * fmaxf(v, 0.f)); break;
          }
        }
      }
    }
  }
}

// ---------------------------------------------------------------------------
// Elementwise / row kernels (wave-per-row, wave32)
// ---------------------------------------------------------------------------
__global__ void hgnn_f32_to_bf16(const float* __restrict__ in, unsigned short* __restrict__ out,
                                 size_t n) {
  size_t i = (size_t)blockIdx.x * blockDim.x + threadIdx.x;
  if (i < n) out[i] = f2bf(in[i]);
}

// x_hyp = proj(expmap0(f1)); store bf16 x_hyp and per-row norm xn
__global__ void hgnn_hyp_in(const float* __restrict__ f1, unsigned short* __restrict__ xh,
                            float* __restrict__ xn, int Nrows) {
  int wid = threadIdx.x >> 5, lane = threadIdx.x & 31;
  int r = blockIdx.x * 8 + wid;
  if (r >= Nrows) return;
  const float* src = f1 + (size_t)r * 512;
  float v[16]; float ss = 0.f;
#pragma unroll
  for (int t = 0; t < 16; ++t) { v[t] = src[lane + 32 * t]; ss += v[t] * v[t]; }
  ss = waveSum(ss);
  float n  = fmaxf(sqrtf(ss), MINF);
  float th = tanhf(n);
  float sc = th / n;
  float nn = th;
  if (nn > PROJ_MAX) { sc *= PROJ_MAX / nn; nn = PROJ_MAX; }
  if (lane == 0) xn[r] = fmaxf(nn, MINF);
  unsigned short* dst = xh + (size_t)r * 512;
#pragma unroll
  for (int t = 0; t < 16; ++t) dst[lane + 32 * t] = f2bf(v[t] * sc);
}

// hb = proj(expmap0(b_hyp_row)) ; single wave, D=256
__global__ void hgnn_expmap_proj_row(const float* __restrict__ src, float* __restrict__ dst) {
  int lane = threadIdx.x & 31;
  float v[8]; float ss = 0.f;
#pragma unroll
  for (int t = 0; t < 8; ++t) { v[t] = src[lane + 32 * t]; ss += v[t] * v[t]; }
  ss = waveSum(ss);
  float n  = fmaxf(sqrtf(ss), MINF);
  float th = tanhf(n);
  float sc = th / n;
  if (th > PROJ_MAX) sc *= PROJ_MAX / th;
#pragma unroll
  for (int t = 0; t < 8; ++t) dst[lane + 32 * t] = v[t] * sc;
}

// xt = logmap0(proj(mobius_add(proj(mobius_matvec_tail(mx)), hb)))
__global__ void hgnn_post_matvec(const float* __restrict__ mx, const float* __restrict__ xn,
                                 const float* __restrict__ hb, float* __restrict__ xt, int Nrows) {
  int wid = threadIdx.x >> 5, lane = threadIdx.x & 31;
  int r = blockIdx.x * 8 + wid;
  if (r >= Nrows) return;
  const float* m = mx + (size_t)r * 256;
  float v[8], hv[8];
  float ss = 0.f, hb2 = 0.f;
#pragma unroll
  for (int t = 0; t < 8; ++t) {
    int e = lane + 32 * t;
    v[t] = m[e]; hv[t] = hb[e];
    ss += v[t] * v[t]; hb2 += hv[t] * hv[t];
  }
  ss = waveSum(ss); hb2 = waveSum(hb2);
  bool allz = (ss == 0.f);
  float mxn = fmaxf(sqrtf(ss), MINF);
  float xr  = xn[r];
  float tt  = tanhf(mxn / xr * artanh_c(xr));
  float sc  = tt / mxn;
  float rn  = allz ? 0.f : tt;
#pragma unroll
  for (int t = 0; t < 8; ++t) v[t] = allz ? 0.f : v[t] * sc;
  if (rn > PROJ_MAX) {
    float f = PROJ_MAX / rn;
#pragma unroll
    for (int t = 0; t < 8; ++t) v[t] *= f;
    rn = PROJ_MAX;
  }
  // mobius_add(v, hb), c = 1
  float x2 = rn * rn;
  float xy = 0.f;
#pragma unroll
  for (int t = 0; t < 8; ++t) xy += v[t] * hv[t];
  xy = waveSum(xy);
  float cx  = 1.f + 2.f * xy + hb2;
  float cy  = 1.f - x2;
  float den = fmaxf(1.f + 2.f * xy + x2 * hb2, MINF);
  float o[8]; float ss2 = 0.f;
#pragma unroll
  for (int t = 0; t < 8; ++t) { o[t] = (cx * v[t] + cy * hv[t]) / den; ss2 += o[t] * o[t]; }
  ss2 = waveSum(ss2);
  float n2 = sqrtf(ss2);
  float nf = n2;
  if (n2 > PROJ_MAX) {
    float f = PROJ_MAX / n2;
#pragma unroll
    for (int t = 0; t < 8; ++t) o[t] *= f;
    nf = PROJ_MAX;
  }
  nf = fmaxf(nf, MINF);
  float lsc = artanh_c(nf) / nf;
  float* out = xt + (size_t)r * 256;
#pragma unroll
  for (int t = 0; t < 8; ++t) out[lane + 32 * t] = o[t] * lsc;
}

// u = logmap0(proj(expmap0(relu(logmap0(proj(expmap0(support))))))) -> bf16
__global__ void hgnn_post_agg(const float* __restrict__ sup, unsigned short* __restrict__ u,
                              int Nrows) {
  int wid = threadIdx.x >> 5, lane = threadIdx.x & 31;
  int r = blockIdx.x * 8 + wid;
  if (r >= Nrows) return;
  const float* s = sup + (size_t)r * 256;
  float v[8]; float ss = 0.f;
#pragma unroll
  for (int t = 0; t < 8; ++t) { v[t] = s[lane + 32 * t]; ss += v[t] * v[t]; }
  ss = waveSum(ss);
  float n  = fmaxf(sqrtf(ss), MINF);
  float th = tanhf(n);
  float sc = th / n;
  float nh = th;
  if (nh > PROJ_MAX) { sc *= PROJ_MAX / nh; nh = PROJ_MAX; }
  float nhc = fmaxf(nh, MINF);
  float tsc = sc * (artanh_c(nhc) / nhc);   // combined expmap0+proj+logmap0 scale
  float tv[8]; float ss2 = 0.f;
#pragma unroll
  for (int t = 0; t < 8; ++t) { tv[t] = fmaxf(v[t] * tsc, 0.f); ss2 += tv[t] * tv[t]; }
  ss2 = waveSum(ss2);
  float nt  = fmaxf(sqrtf(ss2), MINF);
  float th2 = tanhf(nt);
  float sc2 = th2 / nt;
  float n2  = th2;
  if (n2 > PROJ_MAX) { sc2 *= PROJ_MAX / n2; n2 = PROJ_MAX; }
  float n2c = fmaxf(n2, MINF);
  float fin = sc2 * (artanh_c(n2c) / n2c);
  unsigned short* out = u + (size_t)r * 256;
#pragma unroll
  for (int t = 0; t < 8; ++t) out[lane + 32 * t] = f2bf(tv[t] * fin);
}

// one wave per edge: out[dst] += val * xin[src]   (val==nullptr -> 1)
__global__ void hgnn_scatter(const int* __restrict__ src, const int* __restrict__ dst,
                             const float* __restrict__ val, const float* __restrict__ xin,
                             float* __restrict__ out, int E) {
  int gw = (int)(((size_t)blockIdx.x * blockDim.x + threadIdx.x) >> 5);
  int lane = threadIdx.x & 31;
  if (gw >= E) return;
  int s = src[gw], d = dst[gw];
  float v = val ? val[gw] : 1.0f;
  const float* xi = xin + (size_t)s * 256;
  float* o = out + (size_t)d * 256;
#pragma unroll
  for (int t = 0; t < 8; ++t) {
    int j = lane + 32 * t;
    atomicAdd(&o[j], v * xi[j]);
  }
}

// one block per feature: mean / rstd over N rows
__global__ void hgnn_bn_stats(const float* __restrict__ z, float* __restrict__ mu,
                              float* __restrict__ rstd, int Nrows) {
  __shared__ float s1[256], s2[256];
  int f = blockIdx.x;
  float a = 0.f, b = 0.f;
  for (int r = threadIdx.x; r < Nrows; r += blockDim.x) {
    float v = z[(size_t)r * 256 + f];
    a += v; b += v * v;
  }
  s1[threadIdx.x] = a; s2[threadIdx.x] = b;
  __syncthreads();
  for (int st = 128; st > 0; st >>= 1) {
    if (threadIdx.x < st) { s1[threadIdx.x] += s1[threadIdx.x + st]; s2[threadIdx.x] += s2[threadIdx.x + st]; }
    __syncthreads();
  }
  if (threadIdx.x == 0) {
    float m = s1[0] / (float)Nrows;
    mu[f] = m;
    rstd[f] = rsqrtf(s2[0] / (float)Nrows - m * m + 1e-5f);
  }
}

__global__ void hgnn_bn_apply(const float* __restrict__ z, const float* __restrict__ mu,
                              const float* __restrict__ rstd, const float* __restrict__ gamma,
                              const float* __restrict__ beta, unsigned short* __restrict__ zn,
                              size_t total) {
  size_t i = (size_t)blockIdx.x * blockDim.x + threadIdx.x;
  if (i >= total) return;
  int f = (int)(i & 255);
  float v = tanhf((z[i] - mu[f]) * rstd[f] * gamma[f] + beta[f]);
  zn[i] = f2bf(v);
}

__global__ void hgnn_copy_f1(const float* __restrict__ f1, float* __restrict__ x, int Nrows) {
  size_t i = (size_t)blockIdx.x * blockDim.x + threadIdx.x;
  if (i >= (size_t)Nrows * 512) return;
  size_t r = i >> 9;
  int j = (int)(i & 511);
  x[r * 2304 + j] = f1[i];
}

__global__ void hgnn_pair_gather(const float* __restrict__ x, const int* __restrict__ edge_index,
                                 const int* __restrict__ edge_id, unsigned short* __restrict__ xs,
                                 unsigned short* __restrict__ xp, int Bn, int E2v) {
  size_t i = (size_t)blockIdx.x * blockDim.x + threadIdx.x;
  size_t total = (size_t)Bn * 2304;
  if (i >= total) return;
  int b = (int)(i / 2304);
  int j = (int)(i % 2304);
  int e = edge_id[b];
  int i0 = edge_index[e];
  int i1 = edge_index[E2v + e];
  float a = x[(size_t)i0 * 2304 + j];
  float c = x[(size_t)i1 * 2304 + j];
  xs[i] = f2bf(a + c);
  xp[i] = f2bf(a * c);
}

__global__ void hgnn_final(const unsigned short* __restrict__ h2,
                           const unsigned short* __restrict__ Wf3b,
                           const float* __restrict__ bf3, float* __restrict__ out, int Bn) {
  int i = blockIdx.x * blockDim.x + threadIdx.x;
  if (i >= Bn * 7) return;
  int b = i / 7, cc = i % 7;
  const unsigned short* h = h2 + (size_t)b * 576;
  const unsigned short* w = Wf3b + (size_t)cc * 576;
  float acc = bf3[cc];
  for (int k = 0; k < 576; ++k) acc += bf2f(h[k]) * bf2f(w[k]);
  out[i] = acc;
}

// ---------------------------------------------------------------------------
// Host orchestration
// ---------------------------------------------------------------------------
extern "C" void kernel_launch(void* const* d_in, const int* in_sizes, int n_in,
                              void* d_out, int out_size, void* d_ws, size_t ws_size,
                              hipStream_t stream) {
  (void)in_sizes; (void)n_in; (void)out_size; (void)ws_size;
  const int N = 20000, D_IN = 512, D_H = 256, C = 7, E = 320000, E2 = 100000, Bn = 8192, H3 = 2304;

  const float* f1       = (const float*)d_in[0];
  const int*   adj_src  = (const int*)d_in[1];
  const int*   adj_dst  = (const int*)d_in[2];
  const float* adj_val  = (const float*)d_in[3];
  const int*   edge_src = (const int*)d_in[4];
  const int*   edge_dst = (const int*)d_in[5];
  const int*   edge_idx = (const int*)d_in[6];
  const int*   edge_id  = (const int*)d_in[7];
  const float* W_hyp = (const float*)d_in[8];
  const float* b_hyp = (const float*)d_in[9];
  const float* Wd    = (const float*)d_in[10];
  const float* bd    = (const float*)d_in[11];
  const float* Wg1   = (const float*)d_in[12];
  const float* bg1   = (const float*)d_in[13];
  const float* gamma = (const float*)d_in[14];
  const float* beta  = (const float*)d_in[15];
  const float* Wg2   = (const float*)d_in[16];
  const float* bg2   = (const float*)d_in[17];
  const float* Wgate = (const float*)d_in[18];
  const float* bgate = (const float*)d_in[19];
  const float* Wint  = (const float*)d_in[20];
  const float* bint  = (const float*)d_in[21];
  const float* Wout  = (const float*)d_in[22];
  const float* bout  = (const float*)d_in[23];
  const float* Wf1   = (const float*)d_in[24];
  const float* bf1   = (const float*)d_in[25];
  const float* Wf2   = (const float*)d_in[26];
  const float* bf2   = (const float*)d_in[27];
  const float* Wf3   = (const float*)d_in[28];
  const float* bf3   = (const float*)d_in[29];

  // ---- workspace carve ----
  unsigned char* ws = (unsigned char*)d_ws;
  size_t off = 0;
  auto alloc = [&](size_t bytes) -> void* {
    void* p = ws + off;
    off = (off + bytes + 255) & ~(size_t)255;
    return p;
  };
  typedef unsigned short u16;
  u16* W_hyp_b = (u16*)alloc((size_t)C * D_H * D_IN * 2);
  u16* Wd_b    = (u16*)alloc((size_t)C * D_H * D_H * 2);
  u16* Wg1_b   = (u16*)alloc((size_t)C * D_H * D_H * 2);
  u16* Wg2_b   = (u16*)alloc((size_t)C * D_H * D_H * 2);
  u16* Wgate_b = (u16*)alloc((size_t)H3 * H3 * 2);
  u16* Wint_b  = (u16*)alloc((size_t)H3 * H3 * 2);
  u16* Wout_b  = (u16*)alloc((size_t)H3 * H3 * 2);
  u16* Wf1_b   = (u16*)alloc((size_t)(H3 / 2) * H3 * 2);
  u16* Wf2_b   = (u16*)alloc((size_t)(H3 / 4) * (H3 / 2) * 2);
  u16* Wf3_b   = (u16*)alloc((size_t)C * (H3 / 4) * 2);
  u16*  xh_b  = (u16*)alloc((size_t)N * D_IN * 2);
  float* xn   = (float*)alloc((size_t)N * 4);
  float* hb   = (float*)alloc((size_t)D_H * 4);
  float* mx   = (float*)alloc((size_t)N * D_H * 4);
  float* xt   = (float*)alloc((size_t)N * D_H * 4);
  float* sup  = (float*)alloc((size_t)N * D_H * 4);
  u16*  u_b   = (u16*)alloc((size_t)N * D_H * 2);
  float* dbuf = (float*)alloc((size_t)N * D_H * 4);
  float* agg  = (float*)alloc((size_t)N * D_H * 4);
  u16*  agg_b = (u16*)alloc((size_t)N * D_H * 2);
  float* zbuf = (float*)alloc((size_t)N * D_H * 4);
  float* mu   = (float*)alloc(256 * 4);
  float* rstd = (float*)alloc(256 * 4);
  u16*  zn_b  = (u16*)alloc((size_t)N * D_H * 2);
  float* xcat = (float*)alloc((size_t)N * H3 * 4);
  u16*  xs_b  = (u16*)alloc((size_t)Bn * H3 * 2);
  u16*  xp_b  = (u16*)alloc((size_t)Bn * H3 * 2);
  float* gate = (float*)alloc((size_t)Bn * H3 * 4);
  u16*  gi_b  = (u16*)alloc((size_t)Bn * H3 * 2);
  u16*  g_b   = (u16*)alloc((size_t)Bn * H3 * 2);
  u16*  h1_b  = (u16*)alloc((size_t)Bn * (H3 / 2) * 2);
  u16*  h2_b  = (u16*)alloc((size_t)Bn * (H3 / 4) * 2);

  auto conv = [&](const float* src, u16* dst, size_t n) {
    hgnn_f32_to_bf16<<<(unsigned)((n + 255) / 256), 256, 0, stream>>>(src, dst, n);
  };
  auto gemm = [&](const u16* A, const u16* W, const float* bias, int M, int K, int Nout,
                  int epi, float* Cf, u16* Cb, const float* gp, int ldc, int colOff) {
    dim3 g((Nout + TILE_N - 1) / TILE_N, (M + TILE_M - 1) / TILE_M);
    hgnn_gemm_bf16<<<g, 256, 0, stream>>>(A, W, bias, M, K, Nout, epi, Cf, Cb, gp, ldc, colOff);
  };

  // ---- weight conversions (deterministic, every call) ----
  conv(W_hyp, W_hyp_b, (size_t)C * D_H * D_IN);
  conv(Wd, Wd_b, (size_t)C * D_H * D_H);
  conv(Wg1, Wg1_b, (size_t)C * D_H * D_H);
  conv(Wg2, Wg2_b, (size_t)C * D_H * D_H);
  conv(Wgate, Wgate_b, (size_t)H3 * H3);
  conv(Wint, Wint_b, (size_t)H3 * H3);
  conv(Wout, Wout_b, (size_t)H3 * H3);
  conv(Wf1, Wf1_b, (size_t)(H3 / 2) * H3);
  conv(Wf2, Wf2_b, (size_t)(H3 / 4) * (H3 / 2));
  conv(Wf3, Wf3_b, (size_t)C * (H3 / 4));

  // ---- x_hyp = proj(expmap0(f1)) ----
  hgnn_hyp_in<<<(N + 7) / 8, 256, 0, stream>>>(f1, xh_b, xn, N);

  // ---- per-channel graph pipeline ----
  for (int i = 0; i < C; ++i) {
    gemm(xh_b, W_hyp_b + (size_t)i * D_H * D_IN, nullptr, N, D_IN, D_H,
         EPI_F32, mx, nullptr, nullptr, D_H, 0);
    hgnn_expmap_proj_row<<<1, 32, 0, stream>>>(b_hyp + (size_t)i * D_H, hb);
    hgnn_post_matvec<<<(N + 7) / 8, 256, 0, stream>>>(mx, xn, hb, xt, N);
    (void)hipMemsetAsync(sup, 0, (size_t)N * D_H * 4, stream);
    hgnn_scatter<<<E / 8, 256, 0, stream>>>(adj_src + (size_t)i * E, adj_dst + (size_t)i * E,
                                            adj_val + (size_t)i * E, xt, sup, E);
    hgnn_post_agg<<<(N + 7) / 8, 256, 0, stream>>>(sup, u_b, N);
    gemm(u_b, Wd_b + (size_t)i * D_H * D_H, bd + (size_t)i * D_H, N, D_H, D_H,
         EPI_F32, dbuf, nullptr, nullptr, D_H, 0);
    (void)hipMemcpyAsync(agg, dbuf, (size_t)N * D_H * 4, hipMemcpyDeviceToDevice, stream);
    hgnn_scatter<<<E / 8, 256, 0, stream>>>(edge_src + (size_t)i * E, edge_dst + (size_t)i * E,
                                            nullptr, dbuf, agg, E);
    conv(agg, agg_b, (size_t)N * D_H);
    gemm(agg_b, Wg1_b + (size_t)i * D_H * D_H, bg1 + (size_t)i * D_H, N, D_H, D_H,
         EPI_F32, zbuf, nullptr, nullptr, D_H, 0);
    hgnn_bn_stats<<<256, 256, 0, stream>>>(zbuf, mu, rstd, N);
    hgnn_bn_apply<<<(unsigned)(((size_t)N * D_H + 255) / 256), 256, 0, stream>>>(
        zbuf, mu, rstd, gamma + (size_t)i * D_H, beta + (size_t)i * D_H, zn_b, (size_t)N * D_H);
    gemm(zn_b, Wg2_b + (size_t)i * D_H * D_H, bg2 + (size_t)i * D_H, N, D_H, D_H,
         EPI_F32, xcat, nullptr, nullptr, H3, D_IN + i * D_H);
  }

  // ---- concat: x[:, 0:512] = f1 ----
  hgnn_copy_f1<<<(unsigned)(((size_t)N * D_IN + 255) / 256), 256, 0, stream>>>(f1, xcat, N);

  // ---- edge MLP phase ----
  hgnn_pair_gather<<<(unsigned)(((size_t)Bn * H3 + 255) / 256), 256, 0, stream>>>(
      xcat, edge_idx, edge_id, xs_b, xp_b, Bn, E2);
  gemm(xs_b, Wgate_b, bgate, Bn, H3, H3, EPI_F32_SIG, gate, nullptr, nullptr, H3, 0);
  gemm(xp_b, Wint_b, bint, Bn, H3, H3, EPI_BF16_GATE_RELU, nullptr, gi_b, gate, H3, 0);
  gemm(gi_b, Wout_b, bout, Bn, H3, H3, EPI_BF16, nullptr, g_b, nullptr, H3, 0);
  gemm(g_b, Wf1_b, bf1, Bn, H3, H3 / 2, EPI_BF16_RELU, nullptr, h1_b, nullptr, H3 / 2, 0);
  gemm(h1_b, Wf2_b, bf2, Bn, H3 / 2, H3 / 4, EPI_BF16_RELU, nullptr, h2_b, nullptr, H3 / 4, 0);
  hgnn_final<<<(Bn * 7 + 255) / 256, 256, 0, stream>>>(h2_b, Wf3_b, bf3, (float*)d_out, Bn);
}
